// EdgeGNNBlock_43508018708924
// MI455X (gfx1250) — matrix operand order repaired and checked by reading, not compiled, not run on hardware
//
#include <hip/hip_runtime.h>
#include <hip/hip_bf16.h>

#define N_NODES_C   10000
#define N_EDGES_C   320000
#define NODE_C      256
#define K1          768     // 2*NODE_C + IN_EDGE_C
#define OC          512     // OUT_EDGE_C
#define TILE_M      32      // edges per block tile (2 WMMA M-tiles)

typedef __attribute__((ext_vector_type(16))) __bf16 v16bf;
typedef __attribute__((ext_vector_type(8)))  float  v8f;

union ABFrag {
    v16bf v;     // 16 bf16 = 8 VGPRs
    uint4 q[2];  // two 16-byte chunks
};

__device__ __forceinline__ unsigned short bf16_bits(float f) {
    // round-to-nearest-even f32 -> bf16 bit pattern
    unsigned u = __float_as_uint(f);
    unsigned r = u + 0x7FFFu + ((u >> 16) & 1u);
    return (unsigned short)(r >> 16);
}

__device__ __forceinline__ unsigned pack2_bf16(float lo, float hi) {
    return (unsigned)bf16_bits(lo) | ((unsigned)bf16_bits(hi) << 16);
}

// Load 4 B-fragments (one per N-tile) for one 32-wide K step.
__device__ __forceinline__ void load_b4(ABFrag* __restrict__ b,
                                        const unsigned short* __restrict__ w,
                                        int rowbase, int ld, int kb, int hi) {
#pragma unroll
    for (int tn = 0; tn < 4; ++tn) {
        const unsigned short* br = &w[(size_t)(rowbase + tn * 16) * ld + kb + hi * 16];
        b[tn].q[0] = *(const uint4*)br;
        b[tn].q[1] = *(const uint4*)(br + 8);
    }
}

// Load the two A-fragments (M-tiles 0 and 1) for one 32-wide K step from LDS.
__device__ __forceinline__ void load_a2(ABFrag& a0, ABFrag& a1,
                                        const unsigned short* __restrict__ s,
                                        int ld, int kb, int l15, int hi) {
    const unsigned short* ar0 = &s[(l15)      * ld + kb + hi * 16];
    const unsigned short* ar1 = &s[(16 + l15) * ld + kb + hi * 16];
    a0.q[0] = *(const uint4*)ar0;  a0.q[1] = *(const uint4*)(ar0 + 8);
    a1.q[0] = *(const uint4*)ar1;  a1.q[1] = *(const uint4*)(ar1 + 8);
}

// 8 WMMAs: 2 M-tiles x 4 N-tiles against one B quad.
__device__ __forceinline__ void wmma8(v8f acc[2][4], const ABFrag& a0,
                                      const ABFrag& a1, const ABFrag* __restrict__ b) {
#pragma unroll
    for (int tn = 0; tn < 4; ++tn) {
        acc[0][tn] = __builtin_amdgcn_wmma_f32_16x16x32_bf16(
            false, a0.v, false, b[tn].v, (short)0, acc[0][tn], false, false);
        acc[1][tn] = __builtin_amdgcn_wmma_f32_16x16x32_bf16(
            false, a1.v, false, b[tn].v, (short)0, acc[1][tn], false, false);
    }
}

// ---------------------------------------------------------------------------
// Convert W1 [768x512] and W2 [512x512] (row-major f32, (in,out)) to bf16,
// transposed to [N,K] so WMMA B-fragments read contiguous K per lane.
// ---------------------------------------------------------------------------
__global__ void __launch_bounds__(256)
convert_weights(const float* __restrict__ W1, const float* __restrict__ W2,
                unsigned short* __restrict__ w1t, unsigned short* __restrict__ w2t) {
    int idx = blockIdx.x * 256 + threadIdx.x;
    if (idx < K1 * OC) {
        int k = idx / OC;
        int n = idx - k * OC;
        w1t[n * K1 + k] = bf16_bits(W1[idx]);
    } else {
        int j = idx - K1 * OC;
        int k = j / OC;
        int n = j - k * OC;
        w2t[n * OC + k] = bf16_bits(W2[j]);
    }
}

__global__ void __launch_bounds__(256)
zero_out(float* __restrict__ out) {
    size_t i = (size_t)blockIdx.x * 256 + threadIdx.x;
    out[i] = 0.0f;
}

// ---------------------------------------------------------------------------
// Fused per-edge MLP with WMMA bf16 + scatter-add.
// One block = 32-edge tile. 8 waves; wave w owns output columns [64w, 64w+64)
// across BOTH 16-row M-tiles. B stream is software-pipelined one K-step ahead
// in a ping-pong register file so 8 loads overlap 8 WMMAs.
// ---------------------------------------------------------------------------
__global__ void __launch_bounds__(256)
edge_mlp(const float* __restrict__ x,
         const long long* __restrict__ ei,        // [2, E] int64
         const float* __restrict__ ea,            // [E, 256]
         const unsigned short* __restrict__ w1t,  // bf16 [512, 768]  (N,K)
         const unsigned short* __restrict__ w2t,  // bf16 [512, 512]  (N,K)
         const float* __restrict__ b1,
         const float* __restrict__ b2,
         float* __restrict__ out) {               // [N_NODES, 512] accumulator
    // A-staging buffer (32x768 bf16 = 48KB), reused after GEMM1 as the
    // 32x512 bf16 activation buffer (32KB). Barriers separate the lifetimes.
    __shared__ unsigned short smem[TILE_M * K1];
    __shared__ int dst_s[TILE_M];
    __shared__ int src_s[TILE_M];

    const int tid  = threadIdx.x;
    const int lane = tid & 31;
    const int wave = tid >> 5;
    const int hi   = lane >> 4;   // half-wave select
    const int l15  = lane & 15;
    const int rowb = wave * 64 + l15;   // B row base for this lane
    const long long ebase = (long long)blockIdx.x * TILE_M;

    if (tid < TILE_M) {
        src_s[tid] = (int)ei[ebase + tid];
        dst_s[tid] = (int)ei[N_EDGES_C + ebase + tid];
    }
    __syncthreads();

    // ---- stage A = [x_i, x_j - x_i, edge_attr] as bf16 in LDS (float4) ----
    for (int idx = tid; idx < TILE_M * (K1 / 4); idx += 256) {
        const int e  = idx / (K1 / 4);
        const int c  = (idx - e * (K1 / 4)) * 4;
        const int sn = src_s[e];
        const int dn = dst_s[e];
        float4 v;
        if (c < NODE_C) {
            v = *(const float4*)&x[(size_t)dn * NODE_C + c];
        } else if (c < 2 * NODE_C) {
            const int cc = c - NODE_C;
            float4 a = *(const float4*)&x[(size_t)sn * NODE_C + cc];
            float4 b = *(const float4*)&x[(size_t)dn * NODE_C + cc];
            v = make_float4(a.x - b.x, a.y - b.y, a.z - b.z, a.w - b.w);
        } else {
            v = *(const float4*)&ea[(ebase + e) * NODE_C + (c - 2 * NODE_C)];
        }
        uint2 p;
        p.x = pack2_bf16(v.x, v.y);
        p.y = pack2_bf16(v.z, v.w);
        *(uint2*)&smem[e * K1 + c] = p;
    }
    __syncthreads();

    // ---- GEMM1: t = relu(A[32x768] @ W1[768x512] + b1) --------------------
    v8f acc[2][4] = {};
    {
        ABFrag bA[4], bB[4];
        ABFrag a0, a1;
        load_b4(bA, w1t, rowb, K1, 0, hi);               // prologue
        for (int kt = 0; kt < K1 / 32; kt += 2) {
            // step kt: consume bA, prefetch bB for kt+1
            load_b4(bB, w1t, rowb, K1, (kt + 1) * 32, hi);
            load_a2(a0, a1, smem, K1, kt * 32, l15, hi);
            __builtin_prefetch(&w1t[(size_t)rowb * K1 + kt * 32 + 96], 0, 1);
            wmma8(acc, a0, a1, bA);
            // step kt+1: consume bB, prefetch bA for kt+2
            if (kt + 2 < K1 / 32)
                load_b4(bA, w1t, rowb, K1, (kt + 2) * 32, hi);
            load_a2(a0, a1, smem, K1, (kt + 1) * 32, l15, hi);
            wmma8(acc, a0, a1, bB);
        }
    }

    // Issue GEMM2's first B quad now so its latency hides behind the epilogue.
    ABFrag b2A[4], b2B[4];
    load_b4(b2A, w2t, rowb, OC, 0, hi);

    __syncthreads();   // all waves done reading A before overwriting smem with t

    // bias + relu -> bf16 activation buffer (aliases smem)
#pragma unroll
    for (int tn = 0; tn < 4; ++tn) {
        const int n = wave * 64 + tn * 16 + l15;
        const float bias = b1[n];
#pragma unroll
        for (int mt = 0; mt < 2; ++mt) {
#pragma unroll
            for (int r = 0; r < 8; ++r) {
                const int mrow = mt * 16 + r + 8 * hi;  // C/D layout: M = vgpr + 8*(lane>=16)
                float v = acc[mt][tn][r] + bias;
                v = v > 0.0f ? v : 0.0f;
                smem[mrow * OC + n] = bf16_bits(v);
            }
        }
    }
    __syncthreads();

    // ---- GEMM2: h = t[32x512] @ W2[512x512] + b2 --------------------------
    v8f acc2[2][4] = {};
    {
        ABFrag a0, a1;
        for (int kt = 0; kt < OC / 32; kt += 2) {
            load_b4(b2B, w2t, rowb, OC, (kt + 1) * 32, hi);
            load_a2(a0, a1, smem, OC, kt * 32, l15, hi);
            __builtin_prefetch(&w2t[(size_t)rowb * OC + kt * 32 + 96], 0, 1);
            wmma8(acc2, a0, a1, b2A);
            if (kt + 2 < OC / 32)
                load_b4(b2A, w2t, rowb, OC, (kt + 2) * 32, hi);
            load_a2(a0, a1, smem, OC, (kt + 1) * 32, l15, hi);
            wmma8(acc2, a0, a1, b2B);
        }
    }

    // ---- bias + scatter-add into out[dst] ---------------------------------
#pragma unroll
    for (int tn = 0; tn < 4; ++tn) {
        const int n = wave * 64 + tn * 16 + l15;
        const float bias = b2[n];
#pragma unroll
        for (int mt = 0; mt < 2; ++mt) {
#pragma unroll
            for (int r = 0; r < 8; ++r) {
                const int mrow = mt * 16 + r + 8 * hi;
                const float v = acc2[mt][tn][r] + bias;
                atomicAdd(&out[(size_t)dst_s[mrow] * OC + n], v);
            }
        }
    }
}

// ---------------------------------------------------------------------------
// BatchNorm (batch stats, biased var) + ReLU, in place. One block per channel.
// ---------------------------------------------------------------------------
__global__ void __launch_bounds__(256)
bn_relu(float* __restrict__ out, const float* __restrict__ gamma,
        const float* __restrict__ beta) {
    __shared__ float s1[256];
    __shared__ float s2[256];
    const int c   = blockIdx.x;
    const int tid = threadIdx.x;
    float a = 0.0f, b = 0.0f;
    for (int r = tid; r < N_NODES_C; r += 256) {
        float v = out[(size_t)r * OC + c];
        a += v;
        b += v * v;
    }
    s1[tid] = a;
    s2[tid] = b;
    __syncthreads();
    for (int s = 128; s > 0; s >>= 1) {
        if (tid < s) { s1[tid] += s1[tid + s]; s2[tid] += s2[tid + s]; }
        __syncthreads();
    }
    const float mean  = s1[0] * (1.0f / N_NODES_C);
    const float var   = s2[0] * (1.0f / N_NODES_C) - mean * mean;
    const float scale = rsqrtf(var + 1e-5f) * gamma[c];
    const float shift = beta[c];
    for (int r = tid; r < N_NODES_C; r += 256) {
        const size_t i = (size_t)r * OC + c;
        float y = (out[i] - mean) * scale + shift;
        out[i] = y > 0.0f ? y : 0.0f;
    }
}

extern "C" void kernel_launch(void* const* d_in, const int* in_sizes, int n_in,
                              void* d_out, int out_size, void* d_ws, size_t ws_size,
                              hipStream_t stream) {
    const float*     x     = (const float*)d_in[0];
    const long long* ei    = (const long long*)d_in[1];   // int64 edge_index [2,E]
    const float*     ea    = (const float*)d_in[2];
    const float*     W1    = (const float*)d_in[3];
    const float*     b1    = (const float*)d_in[4];
    const float*     W2    = (const float*)d_in[5];
    const float*     b2    = (const float*)d_in[6];
    const float*     gamma = (const float*)d_in[7];
    const float*     beta  = (const float*)d_in[8];
    float* out = (float*)d_out;

    unsigned short* w1t = (unsigned short*)d_ws;            // bf16 [512,768]
    unsigned short* w2t = w1t + (size_t)OC * K1;            // bf16 [512,512]

    convert_weights<<<(K1 * OC + OC * OC) / 256, 256, 0, stream>>>(W1, W2, w1t, w2t);
    zero_out<<<(N_NODES_C * OC) / 256, 256, 0, stream>>>(out);
    edge_mlp<<<N_EDGES_C / TILE_M, 256, 0, stream>>>(x, ei, ea, w1t, w2t, b1, b2, out);
    bn_relu<<<OC, 256, 0, stream>>>(out, gamma, beta);
}
